// HoloLayer_7851200217997
// MI455X (gfx1250) — matrix operand order repaired and checked by reading, not compiled, not run on hardware
//
#include <hip/hip_runtime.h>
#include <cstdint>
#include <cstddef>

// ---------------------------------------------------------------------------
// HoloLayer forward for MI455X (gfx1250, wave32, WMMA)
//   out = x + irfft( cumsum_s( unit(rfft(k)) * rfft(v) ) * conj(unit(rfft(q))) )
// q/k/v = x @ W^T  (three fused bf16-WMMA GEMMs, fp32 accumulate)
// ---------------------------------------------------------------------------

#define PI_F 3.14159265358979323846f
#define EPS_F 1e-8f

#define B_   8
#define S_   2048
#define D_   1024
#define NROWS (B_ * S_)       // 16384
#define NF   513              // rfft bins D/2+1
#define FS   520              // padded spectrum row stride (c32 units)
#define NCH  64               // cumsum chunks along S
#define CHS  (S_ / NCH)       // 32

typedef __attribute__((ext_vector_type(16))) __bf16       v16bf;
typedef __attribute__((ext_vector_type(8)))  float        v8f;
typedef __attribute__((ext_vector_type(4)))  unsigned int u32x4;
typedef __attribute__((ext_vector_type(4)))  float        f32x4;

struct c32 { float x, y; };
__device__ __forceinline__ c32 cadd(c32 a, c32 b) { return {a.x + b.x, a.y + b.y}; }
__device__ __forceinline__ c32 csub(c32 a, c32 b) { return {a.x - b.x, a.y - b.y}; }
__device__ __forceinline__ c32 cmul(c32 a, c32 b) {
  return {a.x * b.x - a.y * b.y, a.x * b.y + a.y * b.x};
}

__device__ __forceinline__ unsigned short f2bfbits(float f) {
  union { float f; unsigned int u; } in; in.f = f;
  unsigned int u = in.u + 0x7FFFu + ((in.u >> 16) & 1u);  // round-to-nearest-even
  return (unsigned short)(u >> 16);
}
__device__ __forceinline__ unsigned int pack2bf(float a, float b) {
  return (unsigned int)f2bfbits(a) | ((unsigned int)f2bfbits(b) << 16);
}

// ---------------------------------------------------------------------------
// Kernel 1: fused QKV GEMM.  C[m,n] = sum_d X[m,d] * W[n,d]  (Linear: x @ W^T)
// Block: 256 thr (8 waves). Block tile 128(M) x 64(N), K-step 32.
// Wave w -> rows [16w,16w+16), 4 col-tiles of 16 -> 3proj x 4 = 12 wmma/K-step.
// Double-buffered LDS tiles (global/WMMA overlap) + register double-buffered
// B fragments (DS-load / matrix-pipe overlap).
// ---------------------------------------------------------------------------
#define TM 128
#define TN 64
#define TK 32
#define NT (D_ / TK)  // 32 K-steps
#define LSTR 40   // padded LDS stride in bf16 (80B: 16B-aligned, conflict-free)

union BF16Frag { v16bf v; u32x4 q[2]; };

// stage one K-step tile set (X: 128x32, W[3]: 64x32) into LDS buffer `nb`
#define STAGE_TILES(nb, kt_)                                                   \
  do {                                                                         \
    _Pragma("unroll")                                                          \
    for (int it_ = 0; it_ < (TM * TK / 4) / 256; ++it_) {                      \
      int idx_ = tid + it_ * 256;                                              \
      int r_ = idx_ >> 3, c4_ = idx_ & 7;                                      \
      f32x4 f4_ = *(const f32x4*)(X + (size_t)(bm + r_) * D_ + (kt_) + c4_*4); \
      unsigned int* d_ = (unsigned int*)&Xs[nb][r_ * LSTR + c4_ * 4];          \
      d_[0] = pack2bf(f4_.x, f4_.y);                                           \
      d_[1] = pack2bf(f4_.z, f4_.w);                                           \
    }                                                                          \
    _Pragma("unroll")                                                          \
    for (int it_ = 0; it_ < (3 * TN * TK / 4) / 256; ++it_) {                  \
      int idx_ = tid + it_ * 256;                                              \
      int p_ = idx_ >> 9;                                                      \
      int rem_ = idx_ & 511;                                                   \
      int r_ = rem_ >> 3, c4_ = rem_ & 7;                                      \
      f32x4 f4_ =                                                              \
          *(const f32x4*)(Wsrc[p_] + (size_t)(bn + r_) * D_ + (kt_) + c4_*4);  \
      unsigned int* d_ = (unsigned int*)&Ws[nb][p_][r_ * LSTR + c4_ * 4];      \
      d_[0] = pack2bf(f4_.x, f4_.y);                                           \
      d_[1] = pack2bf(f4_.z, f4_.w);                                           \
    }                                                                          \
  } while (0)

__global__ __launch_bounds__(256) void qkv_gemm_kernel(
    const float* __restrict__ X,  const float* __restrict__ Wq,
    const float* __restrict__ Wk, const float* __restrict__ Wv,
    float* __restrict__ Qo, float* __restrict__ Ko, float* __restrict__ Vo)
{
  __shared__ alignas(16) __bf16 Xs[2][TM * LSTR];      // 2 x 10.0 KB
  __shared__ alignas(16) __bf16 Ws[2][3][TN * LSTR];   // 2 x 15.0 KB

  const int tid  = threadIdx.x;
  const int wave = tid >> 5;        // 0..7 (wave32)
  const int lane = tid & 31;
  const int m15  = lane & 15;
  const int hh   = (lane >> 4) & 1; // lane-half selects K sub-block (ISA 7.12.2)
  const int bm   = blockIdx.y * TM;
  const int bn   = blockIdx.x * TN;

  const float* Wsrc[3] = { Wq, Wk, Wv };

  v8f acc[3][4];
  const v8f vzero = {0.f, 0.f, 0.f, 0.f, 0.f, 0.f, 0.f, 0.f};
#pragma unroll
  for (int p = 0; p < 3; ++p)
#pragma unroll
    for (int c = 0; c < 4; ++c) acc[p][c] = vzero;

  // prologue: stage K-step 0 into buffer 0
  STAGE_TILES(0, 0);
  __syncthreads();

  for (int t = 0; t < NT; ++t) {
    const int cur = t & 1;

    // stage next tile into the other buffer while this one is consumed
    if (t + 1 < NT) {
      STAGE_TILES(cur ^ 1, (t + 1) * TK);
      // prefetch the tile after that into cache (global_prefetch_b8)
      if (t + 2 < NT)
        __builtin_prefetch(
            (const void*)(X + (size_t)(bm + (tid >> 1)) * D_ + (t + 2) * TK),
            0, 1);
    }

    // ---- A fragment: row 16*wave + m15; VGPR0-7: K=16h..16h+15
    BF16Frag a;
    {
      const __bf16* ap = &Xs[cur][(16 * wave + m15) * LSTR + 16 * hh];
      a.q[0] = *(const u32x4*)(ap);
      a.q[1] = *(const u32x4*)(ap + 8);
    }

    // ---- 12 WMMAs, B fragments double-buffered in registers ----
    BF16Frag bcur;
    {
      const __bf16* bp = &Ws[cur][0][m15 * LSTR + 16 * hh];
      bcur.q[0] = *(const u32x4*)(bp);
      bcur.q[1] = *(const u32x4*)(bp + 8);
    }
#pragma unroll
    for (int i = 0; i < 12; ++i) {
      BF16Frag bnext = bcur;
      if (i < 11) {
        const int c2 = (i + 1) >> 2;        // i+1 = c2*4 + p2 (unused; c-major below)
        const int p2 = (i + 1) - c2 * 4;    // decode: i = c*? -- see mapping
        // mapping: i = c2idx*3 + p2idx  (c-major, 3 projections inner)
        const int ci = (i + 1) / 3;
        const int pi = (i + 1) - ci * 3;
        (void)c2; (void)p2;
        const __bf16* bp = &Ws[cur][pi][(16 * ci + m15) * LSTR + 16 * hh];
        bnext.q[0] = *(const u32x4*)(bp);
        bnext.q[1] = *(const u32x4*)(bp + 8);
      }
      const int c = i / 3;
      const int p = i - c * 3;
      acc[p][c] = __builtin_amdgcn_wmma_f32_16x16x32_bf16(
          false, a.v, false, bcur.v, (short)0, acc[p][c], false, false);
      bcur = bnext;
    }
    __syncthreads();  // next buffer staged AND current buffer reads retired
  }

  // ---- epilogue: C/D layout VGPR r -> M = r + 8h, lane -> N ----
  float* Out[3] = { Qo, Ko, Vo };
#pragma unroll
  for (int p = 0; p < 3; ++p)
#pragma unroll
    for (int c = 0; c < 4; ++c)
#pragma unroll
      for (int r = 0; r < 8; ++r) {
        int m = bm + 16 * wave + r + 8 * hh;
        int n = bn + 16 * c + m15;
        Out[p][(size_t)m * D_ + n] = acc[p][c][r];
      }
}

// ---------------------------------------------------------------------------
// 1024-pt Stockham FFT in LDS. 256 threads, 2 butterflies/thread/stage.
// Twiddle table tw[t] = exp(sign*i*pi*t/512). Result lands back in bufA
// (10 stages = even number of swaps).
// ---------------------------------------------------------------------------
__device__ __forceinline__ void fill_tw(c32* tw, int tid, float sign) {
  for (int t = tid; t < 512; t += 256) {
    float ang = sign * PI_F * (float)t * (1.0f / 512.0f);
    float s, c;
    __sincosf(ang, &s, &c);
    tw[t] = {c, s};
  }
}

__device__ __forceinline__ void fft1024(c32* bufA, c32* bufB, const c32* tw, int tid) {
  c32* src = bufA;
  c32* dst = bufB;
#pragma unroll
  for (int st = 0; st < 10; ++st) {
    const int m = 1 << st;
#pragma unroll
    for (int half = 0; half < 2; ++half) {
      int i  = tid + half * 256;       // butterfly id 0..511
      int k  = i & (m - 1);
      int jm = i - k;                  // j*m
      int j  = i >> st;
      c32 a = src[i];
      c32 b = src[i + 512];            // l*m == 512 invariant
      c32 w = tw[(j << st) & 511];
      dst[jm + i]     = cadd(a, b);    // 2*j*m + k
      dst[jm + i + m] = cmul(w, csub(a, b));
    }
    __syncthreads();
    c32* t = src; src = dst; dst = t;
  }
}

// ---------------------------------------------------------------------------
// Kernel 2: per-row rfft of q,k,v; normalize q,k; emit fqc = conj(unit(fq)),
// p = unit(fk) * fv.  One block per row.
// ---------------------------------------------------------------------------
__global__ __launch_bounds__(256) void fwd_fft_kernel(
    const float* __restrict__ Qi, const float* __restrict__ Ki,
    const float* __restrict__ Vi,
    c32* __restrict__ fqc, c32* __restrict__ p)
{
  __shared__ c32 bufA[1024];
  __shared__ c32 bufB[1024];
  __shared__ c32 fkn[NF];
  __shared__ c32 tw[512];
  const int row = blockIdx.x;
  const int tid = threadIdx.x;
  fill_tw(tw, tid, -1.0f);
  const size_t rbase = (size_t)row * D_;
  const size_t sbase = (size_t)row * FS;

  // ---- q ----
  for (int i = tid; i < D_; i += 256) bufA[i] = { Qi[rbase + i], 0.f };
  __syncthreads();
  fft1024(bufA, bufB, tw, tid);
  for (int i = tid; i < NF; i += 256) {
    c32 f = bufA[i];
    float inv = 1.0f / (sqrtf(f.x * f.x + f.y * f.y) + EPS_F);
    fqc[sbase + i] = { f.x * inv, -f.y * inv };     // conj(unit(fq))
  }
  __syncthreads();

  // ---- k ----
  for (int i = tid; i < D_; i += 256) bufA[i] = { Ki[rbase + i], 0.f };
  __syncthreads();
  fft1024(bufA, bufB, tw, tid);
  for (int i = tid; i < NF; i += 256) {
    c32 f = bufA[i];
    float inv = 1.0f / (sqrtf(f.x * f.x + f.y * f.y) + EPS_F);
    fkn[i] = { f.x * inv, f.y * inv };
  }
  __syncthreads();

  // ---- v ----
  for (int i = tid; i < D_; i += 256) bufA[i] = { Vi[rbase + i], 0.f };
  __syncthreads();
  fft1024(bufA, bufB, tw, tid);
  for (int i = tid; i < NF; i += 256)
    p[sbase + i] = cmul(fkn[i], bufA[i]);
}

// ---------------------------------------------------------------------------
// Kernel 3a/3b/3c: causal cumsum over S, chunked 3-phase scan for parallelism
// (8*513 chains alone would starve HBM). Result written in-place into p:
//   p[b,s,f] <- cumsum_s(p)[b,s,f] * fqc[b,s,f]
// ---------------------------------------------------------------------------
__global__ void chunk_sum_kernel(const c32* __restrict__ p, c32* __restrict__ cs) {
  int f = blockIdx.x * blockDim.x + threadIdx.x;
  if (f >= NF) return;
  int bc = blockIdx.y;                 // b*NCH + chunk
  int b = bc >> 6, ch = bc & (NCH - 1);
  size_t base = (size_t)(b * S_ + ch * CHS) * FS + f;
  c32 acc = {0.f, 0.f};
#pragma unroll 4
  for (int t = 0; t < CHS; ++t) acc = cadd(acc, p[base + (size_t)t * FS]);
  cs[(size_t)bc * FS + f] = acc;
}

__global__ void chunk_prefix_kernel(c32* __restrict__ cs) {
  int f = blockIdx.x * blockDim.x + threadIdx.x;
  if (f >= NF) return;
  int b = blockIdx.y;
  c32 acc = {0.f, 0.f};
  for (int ch = 0; ch < NCH; ++ch) {
    size_t idx = (size_t)(b * NCH + ch) * FS + f;
    c32 t = cs[idx];
    cs[idx] = acc;                     // exclusive prefix
    acc = cadd(acc, t);
  }
}

__global__ void scan_apply_kernel(c32* __restrict__ p, const c32* __restrict__ fqc,
                                  const c32* __restrict__ cs) {
  int f = blockIdx.x * blockDim.x + threadIdx.x;
  if (f >= NF) return;
  int bc = blockIdx.y;
  int b = bc >> 6, ch = bc & (NCH - 1);
  c32 acc = cs[(size_t)bc * FS + f];
  size_t base = (size_t)(b * S_ + ch * CHS) * FS + f;
#pragma unroll 4
  for (int t = 0; t < CHS; ++t) {
    size_t idx = base + (size_t)t * FS;
    acc = cadd(acc, p[idx]);           // inclusive cumsum
    p[idx] = cmul(acc, fqc[idx]);      // * conj(unit(fq)), in place
  }
}

// ---------------------------------------------------------------------------
// Kernel 4: irfft (hermitian-extend 513 -> 1024, inverse Stockham, 1/N) + x
// ---------------------------------------------------------------------------
__global__ __launch_bounds__(256) void irfft_add_kernel(
    const c32* __restrict__ r, const float* __restrict__ X, float* __restrict__ out)
{
  __shared__ c32 bufA[1024];
  __shared__ c32 bufB[1024];
  __shared__ c32 tw[512];
  const int row = blockIdx.x;
  const int tid = threadIdx.x;
  fill_tw(tw, tid, +1.0f);
  const size_t sbase = (size_t)row * FS;
  for (int i = tid; i < NF; i += 256) {
    c32 v = r[sbase + i];
    bufA[i] = v;
    if (i >= 1 && i < 512) bufA[1024 - i] = { v.x, -v.y };
  }
  __syncthreads();
  fft1024(bufA, bufB, tw, tid);
  const float sc = 1.0f / 1024.0f;
  const size_t rbase = (size_t)row * D_;
  for (int i = tid; i < D_; i += 256)
    out[rbase + i] = X[rbase + i] + bufA[i].x * sc;
}

// ---------------------------------------------------------------------------
extern "C" void kernel_launch(void* const* d_in, const int* in_sizes, int n_in,
                              void* d_out, int out_size, void* d_ws, size_t ws_size,
                              hipStream_t stream)
{
  (void)in_sizes; (void)n_in; (void)out_size; (void)ws_size;
  const float* x  = (const float*)d_in[0];
  const float* Wq = (const float*)d_in[1];
  const float* Wk = (const float*)d_in[2];
  const float* Wv = (const float*)d_in[3];
  float* out = (float*)d_out;

  // workspace layout (~330 MB):
  //   q,k,v          : 3 * 16384*1024 f32            = 192 MB
  //   fqc            : 16384*520 c32                 = 68 MB
  //   p (-> result)  : 16384*520 c32                 = 68 MB
  //   cs (chunk sums): 8*64*520 c32                  = 2.1 MB
  const size_t QE = (size_t)NROWS * D_;
  float* q = (float*)d_ws;
  float* k = q + QE;
  float* v = k + QE;
  c32* fqc = (c32*)(v + QE);
  c32* p   = fqc + (size_t)NROWS * FS;
  c32* cs  = p   + (size_t)NROWS * FS;

  dim3 gemmGrid(D_ / TN, NROWS / TM);          // (16, 128)
  qkv_gemm_kernel<<<gemmGrid, 256, 0, stream>>>(x, Wq, Wk, Wv, q, k, v);

  fwd_fft_kernel<<<NROWS, 256, 0, stream>>>(q, k, v, fqc, p);

  dim3 sgrid((NF + 255) / 256, B_ * NCH);      // (3, 512)
  chunk_sum_kernel<<<sgrid, 256, 0, stream>>>(p, cs);
  dim3 pgrid((NF + 255) / 256, B_);            // (3, 8)
  chunk_prefix_kernel<<<pgrid, 256, 0, stream>>>(cs);
  scan_apply_kernel<<<sgrid, 256, 0, stream>>>(p, fqc, cs);

  irfft_add_kernel<<<NROWS, 256, 0, stream>>>(p, x, out);
}
